// SeqRestarter_naligned_6296422056475
// MI455X (gfx1250) — matrix operand order repaired and checked by reading, not compiled, not run on hardware
//
#include <hip/hip_runtime.h>
#include <hip/hip_bf16.h>

typedef unsigned short u16;
typedef unsigned int   u32;
typedef unsigned long long u64;
typedef __attribute__((ext_vector_type(16))) __bf16 v16bf;
typedef __attribute__((ext_vector_type(8)))  float  v8f;
typedef __attribute__((ext_vector_type(4)))  u32    u32x4;
typedef __attribute__((ext_vector_type(8)))  int    i32x8;
typedef __attribute__((ext_vector_type(4)))  int    i32x4;

union V16 { uint4 q[2]; v16bf v; };

#if defined(__has_builtin)
#if __has_builtin(__builtin_amdgcn_tensor_load_to_lds) && __has_builtin(__builtin_amdgcn_s_wait_tensorcnt)
#define HAVE_TDM 1
#endif
#endif
#ifndef HAVE_TDM
#define HAVE_TDM 0
#endif

// ---------------- problem constants ----------------
#define BSZ    8192
#define LSEQ   20
#define NFEAT  172
#define DMODEL 860      // 5*172
#define KPAD   896      // DMODEL padded to multiple of 32
#define NQKV   2580     // 3*DMODEL
#define NQKVP  2624     // padded to multiple of 64
#define DHEAD  430
#define MQKV   (BSZ*LSEQ)   // 163840

// ---------------- helpers ----------------
__device__ __forceinline__ u16 f2bf(float f) {
  union { float f; u32 u; } x; x.f = f;
  u32 u = x.u;
  u32 r = u + 0x7fffu + ((u >> 16) & 1u);   // round-to-nearest-even
  return (u16)(r >> 16);
}
__device__ __forceinline__ float bf2f(u16 h) {
  union { u32 u; float f; } x; x.u = ((u32)h) << 16; return x.f;
}
__device__ __forceinline__ v8f v8zero() {
  v8f z;
#pragma unroll
  for (int i = 0; i < 8; ++i) z[i] = 0.0f;
  return z;
}
__device__ __forceinline__ v8f wmma_bf16(v16bf a, v16bf b, v8f c) {
  // D = A(16x32 bf16) * B(32x16 bf16) + C(16x16 f32)
  return __builtin_amdgcn_wmma_f32_16x16x32_bf16(false, a, false, b, (short)0, c,
                                                 false, false);
}

#if HAVE_TDM
// TDM: DMA a [tile_y x 32-elem] bf16 tile (row stride = stride_elems) from
// global into LDS at lds_byte, inserting 16B of pad after every 64B of data
// (matches the 80B conflict-free LDS row pitch). Tracked with TENSORcnt.
// This toolchain's builtin takes 6 args (clang-23 / therock form):
//   (u32x4 group0, i32x8 group1, i32x4 group2, i32x4 group3, i32x8, i32 cpol)
__device__ __forceinline__ void tdm_load_tile32(const u16* gptr, u32 lds_byte,
                                                int tile_y, u64 stride_elems) {
  u64 ga = (u64)(size_t)gptr;
  u32x4 g0;
  g0[0] = 1u;                                          // count=1 (user D#)
  g0[1] = lds_byte;                                    // lds_addr
  g0[2] = (u32)ga;                                     // global_addr[31:0]
  g0[3] = (u32)((ga >> 32) & 0x01FFFFFFu) | (2u << 30); // addr[56:32] | type=2
  const u32 tdim = 1u << 20;                           // huge dims: no OOB clamp
  i32x8 g1;
  // data_size=1 (2B), pad_enable=1, pad_interval=3 (16 dw), pad_amount=3 (4 dw)
  g1[0] = (int)((1u << 16) | (1u << 20) | (3u << 22) | (3u << 25));
  g1[1] = (int)((tdim & 0xFFFFu) << 16);                       // tensor_dim0 lo16
  g1[2] = (int)(((tdim >> 16) & 0xFFFFu) | ((tdim & 0xFFFFu) << 16)); // d0 hi|d1 lo
  g1[3] = (int)(((tdim >> 16) & 0xFFFFu) | (32u << 16));       // d1 hi | tile_dim0=32
  g1[4] = (int)(u32)tile_y;                                    // tile_dim1 (tile_dim2=0)
  g1[5] = (int)(u32)(stride_elems & 0xFFFFFFFFull);            // dim0 stride lo32
  g1[6] = (int)(u32)((stride_elems >> 32) & 0xFFFFull);        // dim0 stride hi16
  g1[7] = 0;
  i32x4 z4; z4[0] = 0; z4[1] = 0; z4[2] = 0; z4[3] = 0;        // 2-D tensor: groups 2/3 zero
  i32x8 z8;
#pragma unroll
  for (int i = 0; i < 8; ++i) z8[i] = 0;
  __builtin_amdgcn_tensor_load_to_lds(g0, g1, z4, z4, z8, 0);
}
#endif

// ---------------- weight packing: fp32 [srows x scols] -> bf16 [drows x dcols] (zero padded)
__global__ __launch_bounds__(256) void pack_bf16_kernel(
    const float* __restrict__ src, int srows, int scols,
    u16* __restrict__ dst, int drows, int dcols) {
  int idx = blockIdx.x * 256 + threadIdx.x;
  int total = drows * dcols;
  if (idx >= total) return;
  int r = idx / dcols, c = idx - r * dcols;
  float v = (r < srows && c < scols) ? src[(size_t)r * scols + c] : 0.0f;
  dst[idx] = f2bf(v);
}

// ---------------- feature build: gathers + time-cos -> full bf16 [MQKV x KPAD]
// also extracts last_event_feat into xbuf[:,172:860] (bf16) and writes prev_ts.
__global__ __launch_bounds__(256) void build_full_kernel(
    const int* __restrict__ nids, const int* __restrict__ hist_nids,
    const int* __restrict__ aids, const int* __restrict__ eids,
    const float* __restrict__ ts, const int* __restrict__ dirs,
    const float* __restrict__ node_emb, const float* __restrict__ edge_emb,
    const float* __restrict__ anony_emb, const float* __restrict__ time_w,
    const float* __restrict__ time_b,
    u16* __restrict__ full, u16* __restrict__ xbuf, float* __restrict__ out_ts) {
  int b = blockIdx.x / LSEQ;
  int l = blockIdx.x - b * LSEQ;
  int e = b * LSEQ + l;
  int nid = nids[b];
  int hn  = hist_nids[e];
  int d   = dirs[e];
  int sn  = d ? nid : hn;          // src node
  int dn  = d ? hn  : nid;         // dst node
  int aid = aids[e];
  int eid = eids[e];
  float tlast = ts[b * LSEQ + LSEQ - 1];
  float dt    = tlast - ts[e];
  u16* frow = full + (size_t)e * KPAD;
  for (int c = threadIdx.x; c < KPAD; c += 256) {
    float v = 0.0f;
    if (c < DMODEL) {
      int s = c / NFEAT, j = c - s * NFEAT;
      if      (s == 0) v = node_emb[(size_t)sn * NFEAT + j];
      else if (s == 1) v = node_emb[(size_t)dn * NFEAT + j];
      else if (s == 2) v = anony_emb[(size_t)aid * NFEAT + j];
      else if (s == 3) v = edge_emb[(size_t)eid * NFEAT + j];
      else             v = cosf(dt * time_w[j] + time_b[j]);
    }
    u16 bv = f2bf(v);
    if (l == LSEQ - 1) {
      if (c < DMODEL - NFEAT) {                  // first 688 features of last event
        xbuf[(size_t)b * KPAD + NFEAT + c] = bv; // last_event_feat -> fc1 input
        bv = 0;                                  // zeroed in `full` per reference
      }
    }
    frow[c] = bv;
  }
  if (l == LSEQ - 1) {
    if (threadIdx.x < KPAD - DMODEL)             // zero-pad xbuf cols [860,896)
      xbuf[(size_t)b * KPAD + DMODEL + threadIdx.x] = 0;
    if (threadIdx.x == 0) out_ts[b] = tlast;
  }
}

// ---------------- generic tiled bf16 WMMA GEMM: C[MxN] = A[MxK] * B[NxK]^T (+bias)
// A: row-major [M x lda] bf16 (K padded, M multiple of 128)
// B: row-major [Npad x ldb] bf16 (Npad multiple of 64, K padded)
// block tile 128x64, 8 waves, each wave 32x32 (2x2 WMMA tiles).
// K-slabs are double-buffered in LDS and filled by the Tensor Data Mover
// (wave 0 issues descriptors, waits on TENSORcnt) so all waves keep the
// WMMA pipe fed while the DMA engine streams the next slab.
#define LDS_STRIDE 40   // 32 + 8 pad shorts -> 80B rows, conflict-free b128 reads
template <int MODE>
__global__ __launch_bounds__(256) void gemm_bf16_kernel(
    const u16* __restrict__ A, int lda,
    const u16* __restrict__ B, int ldb,
    const float* __restrict__ bias,
    int kchunks, int nmax,
    u16* __restrict__ obf, int ldobf,
    float* __restrict__ of, int ldof) {
  __shared__ u16 sA[2][128 * LDS_STRIDE];
  __shared__ u16 sB[2][64 * LDS_STRIDE];
  const int tid  = threadIdx.x;
  const int bm   = blockIdx.y, bn = blockIdx.x;
  const int wave = tid >> 5, lane = tid & 31;
  const int wm   = wave >> 1, wn = wave & 1;
  const int half = lane >> 4, nl = lane & 15;

  v8f acc[2][2];
  acc[0][0] = v8zero(); acc[0][1] = v8zero();
  acc[1][0] = v8zero(); acc[1][1] = v8zero();

  const u16* Abase = A + (size_t)(bm * 128) * lda;
  const u16* Bbase = B + (size_t)(bn * 64) * ldb;

#if HAVE_TDM
  // flat pointers to __shared__ carry the DS byte offset in addr[31:0]
  const u32 ldsA[2] = { (u32)(size_t)&sA[0][0], (u32)(size_t)&sA[1][0] };
  const u32 ldsB[2] = { (u32)(size_t)&sB[0][0], (u32)(size_t)&sB[1][0] };
  if (wave == 0) {
    tdm_load_tile32(Abase, ldsA[0], 128, (u64)lda);
    tdm_load_tile32(Bbase, ldsB[0], 64, (u64)ldb);
    __builtin_amdgcn_s_wait_tensorcnt(0);
  }
  __syncthreads();
#else
  const int ar = tid >> 2;          // 0..63
  const int ac = (tid & 3) * 8;     // 0,8,16,24
#endif

  int cur = 0;
  for (int kc = 0; kc < kchunks; ++kc) {
    const int k0 = kc * 32;
#if HAVE_TDM
    if (kc + 1 < kchunks && wave == 0) {   // DMA next slab into the other buffer
      tdm_load_tile32(Abase + k0 + 32, ldsA[cur ^ 1], 128, (u64)lda);
      tdm_load_tile32(Bbase + k0 + 32, ldsB[cur ^ 1], 64, (u64)ldb);
    }
#else
    *(uint4*)(&sA[cur][ar * LDS_STRIDE + ac]) =
        *(const uint4*)(Abase + (size_t)ar * lda + k0 + ac);
    *(uint4*)(&sA[cur][(ar + 64) * LDS_STRIDE + ac]) =
        *(const uint4*)(Abase + (size_t)(ar + 64) * lda + k0 + ac);
    *(uint4*)(&sB[cur][ar * LDS_STRIDE + ac]) =
        *(const uint4*)(Bbase + (size_t)ar * ldb + k0 + ac);
    __syncthreads();
#endif

    v16bf afr[2], bfr[2];
#pragma unroll
    for (int ms = 0; ms < 2; ++ms) {
      int m = wm * 32 + ms * 16 + nl;
      V16 u;
      // ISA 16-bit A layout: k = 16*(e>>3) + 8*half + (e&7)
      u.q[0] = *(const uint4*)(&sA[cur][m * LDS_STRIDE + 8 * half]);
      u.q[1] = *(const uint4*)(&sA[cur][m * LDS_STRIDE + 16 + 8 * half]);
      afr[ms] = u.v;
    }
#pragma unroll
    for (int ns = 0; ns < 2; ++ns) {
      int n = wn * 32 + ns * 16 + nl;
      V16 u;
      // ISA 16-bit B layout: k = 16*(e>>3) + 8*half + (e&7)
      u.q[0] = *(const uint4*)(&sB[cur][n * LDS_STRIDE + 8 * half]);
      u.q[1] = *(const uint4*)(&sB[cur][n * LDS_STRIDE + 16 + 8 * half]);
      bfr[ns] = u.v;
    }
#pragma unroll
    for (int ms = 0; ms < 2; ++ms)
#pragma unroll
      for (int ns = 0; ns < 2; ++ns)
        acc[ms][ns] = wmma_bf16(afr[ms], bfr[ns], acc[ms][ns]);

#if HAVE_TDM
    if (wave == 0) __builtin_amdgcn_s_wait_tensorcnt(0);
    __syncthreads();
    cur ^= 1;
#else
    __syncthreads();
#endif
  }

  // epilogue: C layout lane(n = nl, row = r + 8*half)
#pragma unroll
  for (int ms = 0; ms < 2; ++ms) {
#pragma unroll
    for (int ns = 0; ns < 2; ++ns) {
      int n = bn * 64 + wn * 32 + ns * 16 + nl;
#pragma unroll
      for (int r = 0; r < 8; ++r) {
        int m = bm * 128 + wm * 32 + ms * 16 + half * 8 + r;
        float v = acc[ms][ns][r];
        if constexpr (MODE == 0) {           // qkv: bf16 + bias
          if (n < nmax) obf[(size_t)m * ldobf + n] = f2bf(v + bias[n]);
        } else if constexpr (MODE == 1) {    // h_left: f32 out + bf16 into x
          if (n < nmax) {
            float o = v + bias[n];
            of[(size_t)m * ldof + n] = o;
            obf[(size_t)m * ldobf + n] = f2bf(o);
          }
        } else if constexpr (MODE == 2) {    // h = relu(.): bf16, pad cols zeroed
          float o = (n < nmax) ? fmaxf(v + bias[n], 0.0f) : 0.0f;
          obf[(size_t)m * ldobf + n] = f2bf(o);
        } else {                             // h_right: f32 out
          if (n < nmax) of[(size_t)m * ldof + n] = v + bias[n];
        }
      }
    }
  }
}

// ---------------- fused attention + out-proj + mean/relu, one WG per batch row
#define QKV_PITCH 864   // DMODEL padded to mult of 16 for LDS rows
#define CTX_PITCH 904   // KPAD + 8 pad shorts
#define ATTN_SMEM ((3 * LSEQ * QKV_PITCH + 32 * CTX_PITCH) * 2 + 2 * LSEQ * LSEQ * 4)

__global__ __launch_bounds__(128) void attn_kernel(
    const u16* __restrict__ qkv,        // [MQKV x NQKV] bf16
    const int* __restrict__ hist_nids,
    const u16* __restrict__ Wo,         // [KPAD x KPAD] bf16 (out_proj packed)
    const float* __restrict__ ob,       // out_proj_b [860]
    u16* __restrict__ relu_mean) {      // [BSZ x KPAD] bf16
  extern __shared__ char smem[];
  u16* qL   = (u16*)smem;
  u16* kL   = qL + LSEQ * QKV_PITCH;
  u16* vL   = kL + LSEQ * QKV_PITCH;
  u16* ctxL = vL + LSEQ * QKV_PITCH;            // [32 x CTX_PITCH]
  float* sc = (float*)(ctxL + 32 * CTX_PITCH);  // [2][20][20]
  __shared__ float maskv[LSEQ];

  const int b = blockIdx.x, tid = threadIdx.x;
  if (tid < LSEQ)
    maskv[tid] = (hist_nids[b * LSEQ + tid] == 0 && tid != LSEQ - 1) ? -1e30f : 0.0f;

  // stage q,k,v (bf16) into LDS; 860 = 4*215 so 4-col groups never straddle sections
  const u16* base = qkv + (size_t)b * LSEQ * NQKV;
  for (int idx = tid; idx < LSEQ * 645; idx += 128) {
    int row = idx / 645, g = idx - row * 645;
    int col = g * 4;
    uint2 d = *(const uint2*)(base + (size_t)row * NQKV + col);
    int s = col / DMODEL, c = col - s * DMODEL;
    u16* dst = (s == 0 ? qL : (s == 1 ? kL : vL)) + row * QKV_PITCH + c;
    *(uint2*)dst = d;
  }
  // zero ctx padding: rows [20,32) cols [0,896), rows [0,20) cols [860,896)
  for (int idx = tid; idx < 12 * KPAD; idx += 128) {
    int r = 20 + idx / KPAD, c = idx % KPAD;
    ctxL[r * CTX_PITCH + c] = 0;
  }
  for (int idx = tid; idx < LSEQ * (KPAD - DMODEL); idx += 128) {
    int r = idx / (KPAD - DMODEL), c = DMODEL + idx % (KPAD - DMODEL);
    ctxL[r * CTX_PITCH + c] = 0;
  }
  __syncthreads();

  // scores[h][i][j] = q_i . k_j / sqrt(DH)  (masked keys -> -inf)
  const float scale = 0.0482242753f;  // 1/sqrt(430)
  for (int idx = tid; idx < 2 * LSEQ * LSEQ; idx += 128) {
    int h = idx / (LSEQ * LSEQ);
    int rem = idx - h * LSEQ * LSEQ;
    int i = rem / LSEQ, j = rem - i * LSEQ;
    float s;
    if (maskv[j] < -1.0f) {
      s = -1e30f;
    } else {
      float accv = 0.0f;
      const u16* qp = qL + i * QKV_PITCH + h * DHEAD;
      const u16* kp = kL + j * QKV_PITCH + h * DHEAD;
      for (int d0 = 0; d0 < DHEAD; d0 += 2) {
        u32 qw = *(const u32*)(qp + d0);
        u32 kw = *(const u32*)(kp + d0);
        accv += bf2f((u16)qw) * bf2f((u16)kw) +
                bf2f((u16)(qw >> 16)) * bf2f((u16)(kw >> 16));
      }
      s = accv * scale;
    }
    sc[idx] = s;
  }
  __syncthreads();

  // softmax over keys (key 19 always unmasked -> finite)
  if (tid < 2 * LSEQ) {
    int h = tid / LSEQ, i = tid - h * LSEQ;
    float* row = sc + h * LSEQ * LSEQ + i * LSEQ;
    float mx = -1e38f;
    for (int j = 0; j < LSEQ; ++j) mx = fmaxf(mx, row[j]);
    float sum = 0.0f;
    for (int j = 0; j < LSEQ; ++j) { float e = __expf(row[j] - mx); row[j] = e; sum += e; }
    float inv = 1.0f / sum;
    for (int j = 0; j < LSEQ; ++j) row[j] *= inv;
  }
  __syncthreads();

  // ctx[i][c] = sum_j attn[h][i][j] * v[j][c]
  for (int idx = tid; idx < LSEQ * DMODEL; idx += 128) {
    int i = idx / DMODEL, c = idx - i * DMODEL;
    int h = c / DHEAD;
    const float* ar = sc + h * LSEQ * LSEQ + i * LSEQ;
    float accv = 0.0f;
    for (int j = 0; j < LSEQ; ++j) accv += ar[j] * bf2f(vL[j * QKV_PITCH + c]);
    ctxL[i * CTX_PITCH + c] = f2bf(accv);
  }
  __syncthreads();

  // out-proj via WMMA fused with mean-over-L + bias + relu
  const int wave = tid >> 5, lane = tid & 31, half = lane >> 4, nl = lane & 15;
  for (int nt = wave; nt < 54; nt += 4) {   // 54*16 = 864 >= 860 output cols
    v8f a0 = v8zero(), a1 = v8zero();
    const u16* brow = Wo + (size_t)(nt * 16 + nl) * KPAD;
    for (int kc = 0; kc < KPAD / 32; ++kc) {
      const int k0 = kc * 32;
      V16 ua, ub;
      ua.q[0] = *(const uint4*)(ctxL + nl * CTX_PITCH + k0 + 8 * half);
      ua.q[1] = *(const uint4*)(ctxL + nl * CTX_PITCH + k0 + 16 + 8 * half);
      v16bf af0 = ua.v;
      ua.q[0] = *(const uint4*)(ctxL + (16 + nl) * CTX_PITCH + k0 + 8 * half);
      ua.q[1] = *(const uint4*)(ctxL + (16 + nl) * CTX_PITCH + k0 + 16 + 8 * half);
      v16bf af1 = ua.v;
      ub.q[0] = *(const uint4*)(brow + k0 + 8 * half);
      ub.q[1] = *(const uint4*)(brow + k0 + 16 + 8 * half);
      v16bf bf = ub.v;
      a0 = wmma_bf16(af0, bf, a0);   // out rows 0..15
      a1 = wmma_bf16(af1, bf, a1);   // out rows 16..31 (20..31 are zero)
    }
    // column sum over rows held by this lane; combine halves (rows r+8*half)
    float csum = 0.0f;
#pragma unroll
    for (int r = 0; r < 8; ++r) csum += a0[r] + a1[r];
    csum += __shfl_xor(csum, 16, 32);
    int n = nt * 16 + nl;
    if (half == 0) {
      float o = (n < DMODEL) ? fmaxf(csum * (1.0f / LSEQ) + ob[n], 0.0f) : 0.0f;
      relu_mean[(size_t)b * KPAD + n] = f2bf(o);
    }
  }
  if (tid < KPAD - 864)  // zero cols [864,896)
    relu_mean[(size_t)b * KPAD + 864 + tid] = 0;
}

// ---------------- launcher ----------------
static inline size_t rup(size_t x) { return (x + 255) & ~(size_t)255; }

extern "C" void kernel_launch(void* const* d_in, const int* in_sizes, int n_in,
                              void* d_out, int out_size, void* d_ws, size_t ws_size,
                              hipStream_t stream) {
  const int*   nids      = (const int*)d_in[0];
  const int*   hist_nids = (const int*)d_in[1];
  const int*   aids      = (const int*)d_in[2];
  const int*   eids      = (const int*)d_in[3];
  const float* hist_ts   = (const float*)d_in[4];
  const int*   dirs      = (const int*)d_in[5];
  const float* node_emb  = (const float*)d_in[6];
  const float* edge_emb  = (const float*)d_in[7];
  const float* anony_emb = (const float*)d_in[8];
  const float* time_w    = (const float*)d_in[9];
  const float* time_b    = (const float*)d_in[10];
  const float* in_proj_w = (const float*)d_in[11];
  const float* in_proj_b = (const float*)d_in[12];
  const float* out_proj_w= (const float*)d_in[13];
  const float* out_proj_b= (const float*)d_in[14];
  const float* outfn_w   = (const float*)d_in[15];
  const float* outfn_b   = (const float*)d_in[16];
  const float* fc1_w     = (const float*)d_in[17];
  const float* fc1_b     = (const float*)d_in[18];
  const float* fc2_w     = (const float*)d_in[19];
  const float* fc2_b     = (const float*)d_in[20];
  float* out = (float*)d_out;

  // workspace carve (bf16 buffers)
  char* ws = (char*)d_ws;
  size_t off = 0;
  auto carve = [&](size_t bytes) { char* p = ws + off; off += rup(bytes); return p; };
  u16* full      = (u16*)carve((size_t)MQKV * KPAD * 2);       // 293.6 MB
  u16* qkv       = (u16*)carve((size_t)MQKV * NQKV * 2);       // 845.4 MB
  u16* Wq        = (u16*)carve((size_t)NQKVP * KPAD * 2);      // in_proj packed
  u16* Wo        = (u16*)carve((size_t)KPAD * KPAD * 2);       // out_proj packed
  u16* Wfn       = (u16*)carve((size_t)192 * KPAD * 2);        // outfn packed
  u16* Wfc1      = (u16*)carve((size_t)192 * KPAD * 2);
  u16* Wfc2      = (u16*)carve((size_t)192 * 192 * 2);
  u16* relu_mean = (u16*)carve((size_t)BSZ * KPAD * 2);
  u16* xbuf      = (u16*)carve((size_t)BSZ * KPAD * 2);        // [h_left | last_feat]
  u16* hbuf      = (u16*)carve((size_t)BSZ * 192 * 2);
  (void)ws_size; (void)in_sizes; (void)n_in; (void)out_size;

  float* out_left  = out;                      // [8192 x 172]
  float* out_right = out + (size_t)BSZ * NFEAT;
  float* out_ts    = out + (size_t)2 * BSZ * NFEAT;

  // 1) pack weights to bf16 padded layouts
  auto packN = [&](const float* src, int sr, int sc, u16* dst, int dr, int dc) {
    int total = dr * dc;
    pack_bf16_kernel<<<(total + 255) / 256, 256, 0, stream>>>(src, sr, sc, dst, dr, dc);
  };
  packN(in_proj_w,  NQKV,  DMODEL, Wq,   NQKVP, KPAD);
  packN(out_proj_w, DMODEL,DMODEL, Wo,   KPAD,  KPAD);
  packN(outfn_w,    NFEAT, DMODEL, Wfn,  192,   KPAD);
  packN(fc1_w,      NFEAT, DMODEL, Wfc1, 192,   KPAD);
  packN(fc2_w,      NFEAT, NFEAT,  Wfc2, 192,   192);

  // 2) build features (gather + cos), last_event_feat -> xbuf, prev_ts -> d_out
  build_full_kernel<<<MQKV, 256, 0, stream>>>(
      nids, hist_nids, aids, eids, hist_ts, dirs,
      node_emb, edge_emb, anony_emb, time_w, time_b,
      full, xbuf, out_ts);

  // 3) qkv = full @ in_proj_w^T + b   (M=163840, K=896, N=2580)
  gemm_bf16_kernel<0><<<dim3(NQKVP / 64, MQKV / 128), 256, 0, stream>>>(
      full, KPAD, Wq, KPAD, in_proj_b, KPAD / 32, NQKV, qkv, NQKV, nullptr, 0);

  // 4) attention + out-proj + mean/relu (one WG per batch element)
  attn_kernel<<<BSZ, 128, ATTN_SMEM, stream>>>(
      qkv, hist_nids, Wo, out_proj_b, relu_mean);

  // 5) h_left = relu_mean @ outfn_w^T + b  -> f32 out + bf16 into xbuf[:,0:172]
  gemm_bf16_kernel<1><<<dim3(3, BSZ / 128), 256, 0, stream>>>(
      relu_mean, KPAD, Wfn, KPAD, outfn_b, KPAD / 32, NFEAT,
      xbuf, KPAD, out_left, NFEAT);

  // 6) h = relu(x @ fc1_w^T + b) -> bf16 hbuf [8192 x 192]
  gemm_bf16_kernel<2><<<dim3(3, BSZ / 128), 256, 0, stream>>>(
      xbuf, KPAD, Wfc1, KPAD, fc1_b, KPAD / 32, NFEAT,
      hbuf, 192, nullptr, 0);

  // 7) h_right = h @ fc2_w^T + b -> f32 out
  gemm_bf16_kernel<3><<<dim3(3, BSZ / 128), 256, 0, stream>>>(
      hbuf, 192, Wfc2, 192, fc2_b, 192 / 32, NFEAT,
      nullptr, 0, out_right, NFEAT);
}
